// QRNNModel_84774064488817
// MI455X (gfx1250) — matrix operand order repaired and checked by reading, not compiled, not run on hardware
//
#include <hip/hip_runtime.h>
#include <cstdint>
#include <cstddef>

// ---------------------------------------------------------------------------
// QRNN (2-layer, window=1, fo-pooling) for MI455X / gfx1250, wave32 + WMMA.
//   B=128, S=1024, D=64, H=512, gates = 3H = 1536
// Pipeline:
//   (1) convert x -> bf16; transpose+convert W1,W2,Wfc -> bf16 (N,K)-major
//   (2) WMMA GEMM  x@W1+b1  fused tanh/sigmoid -> Z/F/O planes (bf16)
//   (3) fo-pool scan over S -> h1 = sigmoid(o)*c   (bf16, [b,s,h])
//   (4) WMMA GEMM  h1@W2+b2 fused epilogue -> Z/F/O planes (reused buffers)
//   (5) fo-pool scan, keep only final timestep -> h2_last (128x512 bf16)
//   (6) WMMA GEMM  h2_last@Wfc+bfc -> d_out fp32 (128x64)
// GEMM: block = 4 waves = 128x64 tile. The 64xK B (weight) tile is staged into
// LDS ONCE per block via the Tensor Data Mover (TENSOR_LOAD_TO_LDS, TENSORcnt),
// with D# row padding of 16B per K-row -> conflict-free ds_load_b128 reads.
// Each wave: 2 A-frags x 4 B-frags -> 8 WMMA accumulators per k-step.
// ---------------------------------------------------------------------------

typedef __attribute__((ext_vector_type(16))) __bf16 v16bf;
typedef __attribute__((ext_vector_type(8)))  __bf16 v8bf;
typedef __attribute__((ext_vector_type(8)))  float  v8f;
typedef __attribute__((ext_vector_type(4)))  unsigned int v4u;
typedef __attribute__((ext_vector_type(8)))  int v8i;
typedef __attribute__((ext_vector_type(4)))  int v4i;

#define B_   128
#define S_   1024
#define D_   64
#define H_   512
#define NG_  1536        // 3*H
#define M1_  (B_ * S_)   // 131072 rows of (b,s)

#if __has_builtin(__builtin_amdgcn_tensor_load_to_lds)
#define HAVE_TDM 1
#else
#define HAVE_TDM 0
#endif

static __device__ __forceinline__ void wait_tensorcnt0() {
#if __has_builtin(__builtin_amdgcn_s_wait_tensorcnt)
  __builtin_amdgcn_s_wait_tensorcnt(0);
#else
  asm volatile("s_wait_tensorcnt 0x0" ::: "memory");
#endif
}

static __device__ __forceinline__ float fast_sigmoid(float x) {
  // 1/(1+e^-x): v_exp_f32 + v_rcp_f32, no IEEE divide expansion.
  return __builtin_amdgcn_rcpf(1.0f + __expf(-x));
}

static __device__ __forceinline__ float fast_tanh(float x) {
#if __has_builtin(__builtin_amdgcn_tanhf)
  return __builtin_amdgcn_tanhf(x);       // gfx1250 v_tanh_f32
#else
  return 2.0f * fast_sigmoid(2.0f * x) - 1.0f;
#endif
}

// A fragment for v_wmma_f32_16x16x32_bf16 (from global).
// Lane l: row m = base + (l&15); kh = l>>4.
// VGPR0-3 hold K = kb + 8*kh + [0..7]; VGPR4-7 hold K = kb + 16 + 8*kh + [0..7].
static __device__ __forceinline__ v16bf
load_a_frag(const __bf16* __restrict__ A, size_t ldk, size_t mBase, int kb, int lane) {
  const size_t m  = mBase + (lane & 15);
  const int    kh = lane >> 4;
  const __bf16* p = A + m * ldk + (size_t)kb;
  v8bf lo = *(const v8bf*)(p + 8 * kh);
  v8bf hi = *(const v8bf*)(p + 16 + 8 * kh);
  return __builtin_shufflevector(lo, hi, 0,1,2,3,4,5,6,7,8,9,10,11,12,13,14,15);
}

// B fragment from LDS-staged, padded tile (pitch elements = KTOT+8).
// Lane l: col n = nLocal + (l&15); holds K = kb + 8*(l>>4) + [0..7], kb + 16 + 8*(l>>4) + [0..7].
template <int PITCH>
static __device__ __forceinline__ v16bf
load_b_frag_lds(const __bf16* ldsB, int nLocal, int kb, int lane) {
  const int n  = nLocal + (lane & 15);
  const int kh = lane >> 4;
  const __bf16* p = ldsB + n * PITCH + kb + 8 * kh;
  v8bf lo = *(const v8bf*)(p);       // ds_load_b128
  v8bf hi = *(const v8bf*)(p + 16);  // ds_load_b128
  return __builtin_shufflevector(lo, hi, 0,1,2,3,4,5,6,7,8,9,10,11,12,13,14,15);
}

// One wave computes a 32(M) x 64(N) strip: 2 A-fragments x 4 B-fragments
// -> 8 WMMA accumulators. Block = 4 waves -> 128x64 tile; the 64xKTOT weight
// tile is shared by all 4 waves from LDS. Grids divide M,N exactly (no bounds
// checks, EXEC all 1s as WMMA requires).
template <int KTOT, bool GATE>
__global__ void __launch_bounds__(128)
wmma_gemm_kernel(const __bf16* __restrict__ A,    // M x KTOT bf16
                 const __bf16* __restrict__ Bt,   // N x KTOT bf16 (transposed W)
                 const float*  __restrict__ bias, // N
                 __bf16* __restrict__ Zg, __bf16* __restrict__ Fg,
                 __bf16* __restrict__ Og,
                 float* __restrict__ Cout, int N) {
  constexpr int PITCH = KTOT + 8;  // +16B per row: bank-conflict-free b128 reads
  __shared__ __align__(16) __bf16 ldsB[64 * PITCH];

  const int    lane  = threadIdx.x & 31;
  const int    wave  = threadIdx.x >> 5;
  const size_t mBase = (size_t)blockIdx.x * 128 + (size_t)wave * 32;
  const int    nBase = blockIdx.y * 64;

  // ---- stage the 64 x KTOT weight tile into LDS --------------------------
#if HAVE_TDM
  if (wave == 0) {
    // Tensor DMA descriptor (ISA 8.3-8.5): 2D, data_size=2B, tile 64 x KTOT,
    // LDS pad of 4 DWORDs every KTOT*2 bytes -> pitch KTOT+8 elements.
    const unsigned long long ga =
        (unsigned long long)(uintptr_t)(Bt + (size_t)nBase * KTOT);
    const unsigned ldsAddr = (unsigned)(uintptr_t)&ldsB[0];  // low 32 = LDS off
    constexpr unsigned IC = (unsigned)(__builtin_ctz((unsigned)KTOT) - 2);

    v4u g0;
    g0.x = 1u;                                   // count=1, user mode
    g0.y = ldsAddr;                              // lds_addr
    g0.z = (unsigned)(ga & 0xFFFFFFFFu);         // global_addr[31:0]
    g0.w = (unsigned)((ga >> 32) & 0x1FFFFFFu)   // global_addr[56:32]
           | (2u << 30);                         // type = 2 ("image")
    v8i g1;
    g1[0] = (int)((1u << 16)                     // data_size = 1 -> 2 bytes
                  | (1u << 20)                   // pad_enable
                  | (IC << 22)                   // pad_interval: 8<<IC bytes
                  | (3u << 25));                 // pad_amount: 4 DWORDs (16B)
    g1[1] = (int)(((unsigned)KTOT & 0xFFFFu) << 16);  // tensor_dim0[15:0]
    g1[2] = (int)(64u << 16);                    // tensor_dim1 = 64 rows
    g1[3] = (int)((unsigned)KTOT << 16);         // tile_dim0 = KTOT
    g1[4] = 64;                                  // tile_dim1 = 64
    g1[5] = KTOT;                                // tensor_dim0_stride[31:0]
    g1[6] = 0;
    g1[7] = 0;
    v4i g2 = {0, 0, 0, 0};                       // 2D: groups 2/3 unused
    v4i g3 = {0, 0, 0, 0};
    v8i g4 = {0, 0, 0, 0, 0, 0, 0, 0};           // 6-arg toolchain extra group
    __builtin_amdgcn_tensor_load_to_lds(g0, g1, g2, g3, g4, 0);
    wait_tensorcnt0();
  }
#else
  // Fallback: cooperative 16B-chunk copy (global_load_b128 + ds_store_b128).
  for (int c = threadIdx.x; c < 8 * KTOT; c += 128) {
    const int n  = c / (KTOT / 8);
    const int kc = c - n * (KTOT / 8);
    *(v8bf*)&ldsB[n * PITCH + kc * 8] =
        *(const v8bf*)(Bt + (size_t)(nBase + n) * KTOT + kc * 8);
  }
#endif
  __syncthreads();

  // ---- WMMA mainloop: A from global, B from LDS --------------------------
  v8f acc[2][4];
#pragma unroll
  for (int t = 0; t < 2; ++t)
#pragma unroll
    for (int j = 0; j < 4; ++j)
      acc[t][j] = (v8f){0.f, 0.f, 0.f, 0.f, 0.f, 0.f, 0.f, 0.f};

#pragma unroll 4
  for (int kb = 0; kb < KTOT; kb += 32) {
    if (kb + 32 < KTOT) {  // prefetch next A chunks (global_prefetch_b8)
      const size_t m = mBase + (lane & 15);
      __builtin_prefetch(A + m * (size_t)KTOT + kb + 32, 0, 0);
      __builtin_prefetch(A + (m + 16) * (size_t)KTOT + kb + 32, 0, 0);
    }
    v16bf a0 = load_a_frag(A, KTOT, mBase, kb, lane);
    v16bf a1 = load_a_frag(A, KTOT, mBase + 16, kb, lane);
#pragma unroll
    for (int j = 0; j < 4; ++j) {
      v16bf b = load_b_frag_lds<PITCH>(ldsB, j * 16, kb, lane);
      acc[0][j] = __builtin_amdgcn_wmma_f32_16x16x32_bf16(
          false, a0, false, b, (short)0, acc[0][j], false, false);
      acc[1][j] = __builtin_amdgcn_wmma_f32_16x16x32_bf16(
          false, a1, false, b, (short)0, acc[1][j], false, false);
    }
  }

  // ---- epilogue. C/D layout: VGPR r -> row (r + 8*(lane>>4)), col lane&15.
  const int nl   = lane & 15;
  const int half = lane >> 4;
#pragma unroll
  for (int t = 0; t < 2; ++t) {
#pragma unroll
    for (int j = 0; j < 4; ++j) {
      const int   n  = nBase + j * 16 + nl;
      const float bv = bias[n];
#pragma unroll
      for (int r = 0; r < 8; ++r) {
        const size_t m = mBase + t * 16 + 8 * half + r;
        const float  g = acc[t][j][r] + bv;
        if (GATE) {
          const int    plane = n >> 9;          // /H_ ; uniform per 16-wide tile
          const int    h     = n & (H_ - 1);
          const size_t idx   = m * (size_t)H_ + h;
          if (plane == 0)      Zg[idx] = (__bf16)fast_tanh(g);
          else if (plane == 1) Fg[idx] = (__bf16)fast_sigmoid(g);
          else                 Og[idx] = (__bf16)fast_sigmoid(g);
        } else {
          Cout[m * (size_t)N + n] = g;
        }
      }
    }
  }
}

// fo-pool: c_t = f_t*c + (1-f_t)*z_t ; h_t = o_t * c_t. One thread per (b,h),
// sequential over s; consecutive lanes -> consecutive h (coalesced b16 loads).
__global__ void __launch_bounds__(256)
fo_pool_full_kernel(const __bf16* __restrict__ Zg, const __bf16* __restrict__ Fg,
                    const __bf16* __restrict__ Og, __bf16* __restrict__ H1) {
  const int    idx  = blockIdx.x * blockDim.x + threadIdx.x;  // 0..B*H-1
  const int    b    = idx >> 9;
  const int    h    = idx & (H_ - 1);
  const size_t base = (size_t)b * S_ * H_ + h;
  float c = 0.f;
  for (int s = 0; s < S_; ++s) {
    const size_t i = base + (size_t)s * H_;
    const float  z = (float)Zg[i];
    const float  f = (float)Fg[i];
    c = f * c + (1.f - f) * z;
    H1[i] = (__bf16)((float)Og[i] * c);
  }
}

// Layer-2 scan: only the final timestep's h is needed.
__global__ void __launch_bounds__(256)
fo_pool_last_kernel(const __bf16* __restrict__ Zg, const __bf16* __restrict__ Fg,
                    const __bf16* __restrict__ Og, __bf16* __restrict__ H2last) {
  const int    idx  = blockIdx.x * blockDim.x + threadIdx.x;
  const int    b    = idx >> 9;
  const int    h    = idx & (H_ - 1);
  const size_t base = (size_t)b * S_ * H_ + h;
  float c = 0.f;
  for (int s = 0; s < S_; ++s) {
    const size_t i = base + (size_t)s * H_;
    const float  z = (float)Zg[i];
    const float  f = (float)Fg[i];
    c = f * c + (1.f - f) * z;
  }
  const size_t il = base + (size_t)(S_ - 1) * H_;
  H2last[(size_t)b * H_ + h] = (__bf16)((float)Og[il] * c);
}

__global__ void __launch_bounds__(256)
cvt_f32_bf16_kernel(const float* __restrict__ in, __bf16* __restrict__ out, int n) {
  const int i = blockIdx.x * blockDim.x + threadIdx.x;
  if (i < n) out[i] = (__bf16)in[i];
}

// in: K x N fp32 (row-major, row=k) -> out: N x K bf16 (row-major, row=n)
__global__ void __launch_bounds__(256)
transpose_cvt_kernel(const float* __restrict__ in, __bf16* __restrict__ out,
                     int K, int N) {
  const int i = blockIdx.x * blockDim.x + threadIdx.x;
  if (i < K * N) {
    const int n = i / K;
    const int k = i - n * K;
    out[(size_t)n * K + k] = (__bf16)in[(size_t)k * N + n];
  }
}

extern "C" void kernel_launch(void* const* d_in, const int* in_sizes, int n_in,
                              void* d_out, int out_size, void* d_ws, size_t ws_size,
                              hipStream_t stream) {
  (void)in_sizes; (void)n_in; (void)out_size; (void)ws_size;
  const float* x   = (const float*)d_in[0];   // (B,S,D)
  const float* W1  = (const float*)d_in[1];   // (D,3H)
  const float* b1  = (const float*)d_in[2];   // (3H)
  const float* W2  = (const float*)d_in[3];   // (H,3H)
  const float* b2  = (const float*)d_in[4];   // (3H)
  const float* Wfc = (const float*)d_in[5];   // (H,D)
  const float* bfc = (const float*)d_in[6];   // (D)
  float* out = (float*)d_out;                 // (B,D) fp32

  // ---- workspace carve-up (256B aligned) ----
  char*  ws  = (char*)d_ws;
  size_t off = 0;
  auto carve = [&](size_t bytes) {
    void* p = ws + off;
    off += (bytes + 255) & ~(size_t)255;
    return p;
  };
  __bf16* Xb    = (__bf16*)carve((size_t)M1_ * D_ * 2);     //  16 MB
  __bf16* W1t   = (__bf16*)carve((size_t)NG_ * D_ * 2);     // 192 KB (N,K)
  __bf16* W2t   = (__bf16*)carve((size_t)NG_ * H_ * 2);     // 1.5 MB (N,K)
  __bf16* Wfct  = (__bf16*)carve((size_t)D_ * H_ * 2);      //  64 KB (N,K)
  __bf16* Zg    = (__bf16*)carve((size_t)M1_ * H_ * 2);     // 134 MB
  __bf16* Fg    = (__bf16*)carve((size_t)M1_ * H_ * 2);     // 134 MB
  __bf16* Og    = (__bf16*)carve((size_t)M1_ * H_ * 2);     // 134 MB
  __bf16* H1    = (__bf16*)carve((size_t)M1_ * H_ * 2);     // 134 MB
  __bf16* H2l   = (__bf16*)carve((size_t)B_ * H_ * 2);      // 128 KB

  // (1) precision conversion / weight transposes
  {
    int n = M1_ * D_;
    cvt_f32_bf16_kernel<<<(n + 255) / 256, 256, 0, stream>>>(x, Xb, n);
  }
  transpose_cvt_kernel<<<(D_ * NG_ + 255) / 256, 256, 0, stream>>>(W1, W1t, D_, NG_);
  transpose_cvt_kernel<<<(H_ * NG_ + 255) / 256, 256, 0, stream>>>(W2, W2t, H_, NG_);
  transpose_cvt_kernel<<<(H_ * D_ + 255) / 256, 256, 0, stream>>>(Wfc, Wfct, H_, D_);

  // (2) layer-1 GEMM + fused gate activations
  wmma_gemm_kernel<D_, true><<<dim3(M1_ / 128, NG_ / 64), 128, 0, stream>>>(
      Xb, W1t, b1, Zg, Fg, Og, nullptr, NG_);

  // (3) layer-1 fo-pool scan -> h1
  fo_pool_full_kernel<<<(B_ * H_) / 256, 256, 0, stream>>>(Zg, Fg, Og, H1);

  // (4) layer-2 GEMM + fused gate activations (gate buffers reused)
  wmma_gemm_kernel<H_, true><<<dim3(M1_ / 128, NG_ / 64), 128, 0, stream>>>(
      H1, W2t, b2, Zg, Fg, Og, nullptr, NG_);

  // (5) layer-2 scan, final timestep only -> h2_last
  fo_pool_last_kernel<<<(B_ * H_) / 256, 256, 0, stream>>>(Zg, Fg, Og, H2l);

  // (6) final FC: (128 x 512) @ (512 x 64) -> d_out fp32
  wmma_gemm_kernel<H_, false><<<dim3(B_ / 128, D_ / 64), 128, 0, stream>>>(
      H2l, Wfct, bfc, nullptr, nullptr, nullptr, out, D_);
}
